// ModelE_44083544326534
// MI455X (gfx1250) — matrix operand order repaired and checked by reading, not compiled, not run on hardware
//
#include <hip/hip_runtime.h>
#include <hip/hip_bf16.h>

// ---------------------------------------------------------------------------
// MI455X (gfx1250) implementation of the point-cloud re-ID network.
// All channel-mixing contractions go through v_wmma_f32_16x16x32_bf16.
// Workspace requirement: ~280 MB (big scratch region time-multiplexed between
// the edge-conv stage and the SA stage; all bulk activations stored bf16).
// ---------------------------------------------------------------------------

#define EDGE_K 20
#define NSAMP  64
#define CDIV(a, b) (((a) + (b) - 1) / (b))

typedef __attribute__((ext_vector_type(8)))  unsigned short us8;
typedef __attribute__((ext_vector_type(16))) unsigned short us16;
typedef __attribute__((ext_vector_type(16))) __bf16         v16bf;
typedef __attribute__((ext_vector_type(8)))  float          v8f;

__device__ __forceinline__ unsigned short f2bf(float f) {
  unsigned int u = __float_as_uint(f);
  unsigned int r = u + 0x7FFFu + ((u >> 16) & 1u);   // round-to-nearest-even
  if ((u & 0x7F800000u) == 0x7F800000u) r = u;       // inf/nan: truncate
  return (unsigned short)(r >> 16);
}
__device__ __forceinline__ float bf2f(unsigned short h) {
  return __uint_as_float(((unsigned int)h) << 16);
}

// ---------------------------------------------------------------------------
// Generic bf16 WMMA GEMM:  D[p, m] = sum_k A[m, k] * B[p, k]  (+ bias[m])
//   A: [M x K] bf16 row-major (weights), lda == K, K % 32 == 0
//   B: [Npts x ldb] bf16 row-major (point-major activations)
//   D: [Npts x ldd] (f32 or bf16, template OM)
// Each wave computes a 16(M) x 32(N) strip: one A-fragment feeds two WMMAs.
// 4 waves/block -> 32(M) x 64(N) per block.
// Out-of-range rows/cols are handled by CLAMPING addresses, not masking:
// D[m,:] depends only on A[m,:] and D[:,n] only on B[:,n], so garbage from a
// clamped (in-bounds) address only reaches outputs the store mask drops.
// This keeps the K-loop straight-line (no EXEC manipulation, clauses form).
// ---------------------------------------------------------------------------
template <int OM> // 0 = f32 out, 1 = bf16 out
__global__ __launch_bounds__(128) void gemm_kernel(
    const unsigned short* __restrict__ A, const unsigned short* __restrict__ Bm,
    float* __restrict__ Df, unsigned short* __restrict__ Dbf,
    const float* __restrict__ bias,
    int M, int Npts, int K, int ldb, int ldd,
    long long strideB, long long strideD) {
  int lane = threadIdx.x & 31;
  int wave = threadIdx.x >> 5;
  int m0 = (blockIdx.x * 2 + (wave >> 1)) * 16;
  int p0 = blockIdx.y * 64 + (wave & 1) * 32;
  if (m0 >= M || p0 >= Npts) return;
  const unsigned short* Bp = Bm + (long long)blockIdx.z * strideB;

  int arow = m0 + (lane & 15);           if (arow >= M) arow = M - 1;
  int akb  = (lane >> 4) * 16;           // A-frag K sub-base (contiguous, matches B)
  int prow0 = p0 + (lane & 15);          if (prow0 >= Npts) prow0 = Npts - 1;
  int prow1 = p0 + 16 + (lane & 15);     if (prow1 >= Npts) prow1 = Npts - 1;
  int bkb  = (lane >> 4) * 16;           // B-frag K sub-base (contiguous, matches A)

  const unsigned short* ap  = A  + (long long)arow  * K   + akb;
  const unsigned short* bp0 = Bp + (long long)prow0 * ldb + bkb;
  const unsigned short* bp1 = Bp + (long long)prow1 * ldb + bkb;

  v8f acc0 = {0.f, 0.f, 0.f, 0.f, 0.f, 0.f, 0.f, 0.f};
  v8f acc1 = {0.f, 0.f, 0.f, 0.f, 0.f, 0.f, 0.f, 0.f};
#pragma unroll 2
  for (int kk = 0; kk < K; kk += 32) {
    us8 lo = *(const us8*)(ap + kk);
    us8 hi = *(const us8*)(ap + kk + 8);
    us16 ub0 = *(const us16*)(bp0 + kk);
    us16 ub1 = *(const us16*)(bp1 + kk);
    if (kk + 32 < K) {
      __builtin_prefetch(bp0 + kk + 32, 0, 1);  // global_prefetch_b8
      __builtin_prefetch(bp1 + kk + 32, 0, 1);
    }
    us16 ua;
#pragma unroll
    for (int i = 0; i < 8; ++i) { ua[i] = lo[i]; ua[i + 8] = hi[i]; }
    v16bf va  = __builtin_bit_cast(v16bf, ua);
    v16bf vb0 = __builtin_bit_cast(v16bf, ub0);
    v16bf vb1 = __builtin_bit_cast(v16bf, ub1);
    acc0 = __builtin_amdgcn_wmma_f32_16x16x32_bf16(
        false, va, false, vb0, (short)0, acc0, false, false);
    acc1 = __builtin_amdgcn_wmma_f32_16x16x32_bf16(
        false, va, false, vb1, (short)0, acc1, false, false);
  }

  int mbase = m0 + (lane >> 4) * 8;      // D layout: vgpr v -> row mbase+v
  int pa = p0 + (lane & 15);
  int pb = pa + 16;
  long long base = (long long)blockIdx.z * strideD + mbase;
  if (pa < Npts) {
    long long dro = base + (long long)pa * ldd;
#pragma unroll
    for (int v = 0; v < 8; ++v) {
      int m = mbase + v;
      if (m < M) {
        float y = acc0[v] + (bias ? bias[m] : 0.f);
        if (OM == 0) Df[dro + v] = y; else Dbf[dro + v] = f2bf(y);
      }
    }
  }
  if (pb < Npts) {
    long long dro = base + (long long)pb * ldd;
#pragma unroll
    for (int v = 0; v < 8; ++v) {
      int m = mbase + v;
      if (m < M) {
        float y = acc1[v] + (bias ? bias[m] : 0.f);
        if (OM == 0) Df[dro + v] = y; else Dbf[dro + v] = f2bf(y);
      }
    }
  }
}

// ----------------------------- utility kernels -----------------------------

__global__ void cast_pad_kernel(const float* __restrict__ src,
                                unsigned short* __restrict__ dst,
                                int R, int Cs, int Cd) {
  long long t = (long long)blockIdx.x * blockDim.x + threadIdx.x;
  long long total = (long long)R * Cd;
  if (t >= total) return;
  int c = (int)(t % Cd);
  long long r = t / Cd;
  dst[t] = (c < Cs) ? f2bf(src[r * Cs + c]) : (unsigned short)0;
}

__global__ void knn_kernel(const float* __restrict__ xyz, int* __restrict__ idx,
                           int B, int N) {
  int t = blockIdx.x * blockDim.x + threadIdx.x;
  if (t >= B * N) return;
  int b = t / N, n = t % N;
  const float* base = xyz + (long long)b * N * 3;
  float qx = base[n * 3], qy = base[n * 3 + 1], qz = base[n * 3 + 2];
  float bd[EDGE_K];
  int   bi[EDGE_K];
#pragma unroll
  for (int i = 0; i < EDGE_K; ++i) { bd[i] = 3.0e38f; bi[i] = 0; }
  for (int m = 0; m < N; ++m) {
    float dx = base[m * 3] - qx, dy = base[m * 3 + 1] - qy, dz = base[m * 3 + 2] - qz;
    float d = dx * dx + dy * dy + dz * dz;
    if (d < bd[EDGE_K - 1]) {
      int j = EDGE_K - 1;
      while (j > 0 && bd[j - 1] > d) { bd[j] = bd[j - 1]; bi[j] = bi[j - 1]; --j; }
      bd[j] = d; bi[j] = m;
    }
  }
  int* o = idx + (long long)t * EDGE_K;
#pragma unroll
  for (int i = 0; i < EDGE_K; ++i) o[i] = bi[i];
}

// Build edge features e[row, :] = [nb - central | central] (bf16, K-padded).
__global__ void edge_build_kernel(const float* __restrict__ h,
                                  const int* __restrict__ idx,
                                  unsigned short* __restrict__ e,
                                  int B, int N, int Cin, int groups, int Kpad) {
  int rows = B * N * EDGE_K;
  int t = blockIdx.x * blockDim.x + threadIdx.x;
  if (t >= rows) return;
  int b = t / (N * EDGE_K);
  int rem = t % (N * EDGE_K);
  int n = rem / EDGE_K;
  int k = rem % EDGE_K;
  int nb = idx[(long long)(b * N + n) * EDGE_K + k];
  const float* hc = h + ((long long)b * N + n) * Cin;
  const float* hn = h + ((long long)b * N + nb) * Cin;
  int L = (groups == 1) ? Kpad : 2 * Kpad;
  unsigned short* row = e + (long long)t * L;
  if (groups == 1) {
    for (int c = 0; c < Cin; ++c)       row[c]       = f2bf(hn[c] - hc[c]);
    for (int c = 0; c < Cin; ++c)       row[Cin + c] = f2bf(hc[c]);
    for (int c = 2 * Cin; c < Kpad; ++c) row[c] = 0;
  } else {
    unsigned short* s0 = row;
    unsigned short* s1 = row + Kpad;
    for (int c = 0; c < Cin; ++c)   s0[c] = f2bf(hn[c] - hc[c]);
    for (int c = Cin; c < Kpad; ++c) s0[c] = 0;
    for (int c = 0; c < Cin; ++c)   s1[c] = f2bf(hc[c]);
    for (int c = Cin; c < Kpad; ++c) s1[c] = 0;
  }
}

// out[b, n, c] = max_k in_bf16[b, n*Kg + k, c]
__global__ void maxk_kernel(const unsigned short* __restrict__ in,
                            float* __restrict__ out, int B, int N, int C, int Kg) {
  long long total = (long long)B * N * C;
  long long t = (long long)blockIdx.x * blockDim.x + threadIdx.x;
  if (t >= total) return;
  int c = (int)(t % C);
  long long bn = t / C;
  const unsigned short* p = in + bn * (long long)Kg * C + c;
  float m = -3.0e38f;
  for (int k = 0; k < Kg; ++k) m = fmaxf(m, bf2f(p[(long long)k * C]));
  out[t] = m;
}

__global__ void bn_stats_kernel(const float* __restrict__ xf,
                                const unsigned short* __restrict__ xb,
                                float* __restrict__ stats, int R, int C) {
  int c = blockIdx.x;
  float s = 0.f, ss = 0.f;
  for (int r = threadIdx.x; r < R; r += blockDim.x) {
    float v = xf ? xf[(long long)r * C + c] : bf2f(xb[(long long)r * C + c]);
    s += v; ss += v * v;
  }
  __shared__ float shs[256], shq[256];
  shs[threadIdx.x] = s; shq[threadIdx.x] = ss;
  __syncthreads();
  for (int o = 128; o > 0; o >>= 1) {
    if (threadIdx.x < o) { shs[threadIdx.x] += shs[threadIdx.x + o];
                           shq[threadIdx.x] += shq[threadIdx.x + o]; }
    __syncthreads();
  }
  if (threadIdx.x == 0) {
    float m = shs[0] / (float)R;
    float v = fmaxf(shq[0] / (float)R - m * m, 0.f);
    stats[2 * c] = m; stats[2 * c + 1] = v;
  }
}

// y = act(bn(x)) (+ residual); writes f32 and/or bf16 mirrors.
__global__ void bn_apply_kernel(const float* __restrict__ xf,
                                const unsigned short* __restrict__ xb,
                                const float* __restrict__ resf,
                                const unsigned short* __restrict__ resb,
                                float* __restrict__ outf,
                                unsigned short* __restrict__ outb,
                                const float* __restrict__ gamma,
                                const float* __restrict__ beta,
                                const float* __restrict__ stats,
                                long long total, int C, int act) {
  long long t = (long long)blockIdx.x * blockDim.x + threadIdx.x;
  if (t >= total) return;
  int c = (int)(t % C);
  float x = xf ? xf[t] : bf2f(xb[t]);
  float m = stats[2 * c], v = stats[2 * c + 1];
  float y = (x - m) * rsqrtf(v + 1e-5f) * gamma[c] + beta[c];
  if (act == 1)      y = y > 0.f ? y : 0.f;
  else if (act == 2) y = y > 0.f ? y : 0.2f * y;
  if (resf) y += resf[t];
  else if (resb) y += bf2f(resb[t]);
  if (outf) outf[t] = y;
  if (outb) outb[t] = f2bf(y);
}

// Farthest point sampling: one block per batch, dist array in LDS.
__global__ __launch_bounds__(256) void fps_kernel(const float* __restrict__ xyz,
                                                  int* __restrict__ fidx,
                                                  float* __restrict__ centers,
                                                  int N, int S) {
  __shared__ float sdist[2048];
  __shared__ float rv[256];
  __shared__ int   ri[256];
  __shared__ int   ssel;
  int b = blockIdx.x;
  const float* pb = xyz + (long long)b * N * 3;
  for (int i = threadIdx.x; i < N; i += 256) sdist[i] = 1e10f;
  if (threadIdx.x == 0) {
    fidx[b * S] = 0;
    centers[(long long)b * S * 3 + 0] = pb[0];
    centers[(long long)b * S * 3 + 1] = pb[1];
    centers[(long long)b * S * 3 + 2] = pb[2];
    ssel = 0;
  }
  __syncthreads();
  int last = 0;
  for (int t = 1; t < S; ++t) {
    float lx = pb[last * 3], ly = pb[last * 3 + 1], lz = pb[last * 3 + 2];
    float best = -1.f; int barg = N;
    for (int i = threadIdx.x; i < N; i += 256) {
      float dx = pb[i * 3] - lx, dy = pb[i * 3 + 1] - ly, dz = pb[i * 3 + 2] - lz;
      float nd = fminf(sdist[i], dx * dx + dy * dy + dz * dz);
      sdist[i] = nd;
      if (nd > best) { best = nd; barg = i; }
    }
    rv[threadIdx.x] = best; ri[threadIdx.x] = barg;
    __syncthreads();
    for (int o = 128; o > 0; o >>= 1) {
      if (threadIdx.x < o) {
        float v2 = rv[threadIdx.x + o]; int i2 = ri[threadIdx.x + o];
        if (v2 > rv[threadIdx.x] ||
            (v2 == rv[threadIdx.x] && i2 < ri[threadIdx.x])) {
          rv[threadIdx.x] = v2; ri[threadIdx.x] = i2;
        }
      }
      __syncthreads();
    }
    if (threadIdx.x == 0) {
      int sel = ri[0];
      ssel = sel;
      fidx[b * S + t] = sel;
      centers[((long long)b * S + t) * 3 + 0] = pb[sel * 3];
      centers[((long long)b * S + t) * 3 + 1] = pb[sel * 3 + 1];
      centers[((long long)b * S + t) * 3 + 2] = pb[sel * 3 + 2];
    }
    __syncthreads();
    last = ssel;
  }
}

__global__ void ball_kernel(const float* __restrict__ xyz,
                            const float* __restrict__ centers,
                            int* __restrict__ gidx, int B, int N, int S) {
  int t = blockIdx.x * blockDim.x + threadIdx.x;
  if (t >= B * S) return;
  int b = t / S;
  const float* pb = xyz + (long long)b * N * 3;
  const float* cp = centers + (long long)t * 3;
  float cx = cp[0], cy = cp[1], cz = cp[2];
  const float r2 = 0.2f * 0.2f;
  int* o = gidx + (long long)t * NSAMP;
  int cnt = 0;
  for (int n = 0; n < N && cnt < NSAMP; ++n) {
    float dx = pb[n * 3] - cx, dy = pb[n * 3 + 1] - cy, dz = pb[n * 3 + 2] - cz;
    if (dx * dx + dy * dy + dz * dz < r2) o[cnt++] = n;
  }
  int fill = (cnt > 0) ? o[0] : 0;
  for (int j = cnt; j < NSAMP; ++j) o[j] = fill;
}

// Build SA input rows: [xyz - center (3) | h[gidx] (C) | zero pad] (bf16).
__global__ void group_build_kernel(const float* __restrict__ xyz,
                                   const float* __restrict__ centers,
                                   const float* __restrict__ h,
                                   const int* __restrict__ gidx,
                                   unsigned short* __restrict__ xsa,
                                   int B, int N, int S, int C, int Kpad) {
  int rows = B * S * NSAMP;
  int t = blockIdx.x * blockDim.x + threadIdx.x;
  if (t >= rows) return;
  int b = t / (S * NSAMP);
  int rem = t % (S * NSAMP);
  int s = rem / NSAMP;
  int g = gidx[(long long)t];
  const float* p  = xyz + ((long long)b * N + g) * 3;
  const float* cp = centers + ((long long)b * S + s) * 3;
  const float* hf = h + ((long long)b * N + g) * C;
  unsigned short* row = xsa + (long long)t * Kpad;
  row[0] = f2bf(p[0] - cp[0]);
  row[1] = f2bf(p[1] - cp[1]);
  row[2] = f2bf(p[2] - cp[2]);
  for (int c = 0; c < C; ++c) row[3 + c] = f2bf(hf[c]);
  for (int c = 3 + C; c < Kpad; ++c) row[c] = 0;
}

__global__ void se_mean_kernel(const float* __restrict__ h, float* __restrict__ sv,
                               int B, int S, int C) {
  int t = blockIdx.x * blockDim.x + threadIdx.x;
  if (t >= B * C) return;
  int b = t / C, c = t % C;
  float s = 0.f;
  for (int i = 0; i < S; ++i) s += h[((long long)b * S + i) * C + c];
  sv[t] = s / (float)S;
}

__global__ void se1_kernel(const float* __restrict__ sv, const float* __restrict__ w1,
                           float* __restrict__ hid, int B, int C, int Ch) {
  int t = blockIdx.x * blockDim.x + threadIdx.x;
  if (t >= B * Ch) return;
  int b = t / Ch, j = t % Ch;
  const float* w = w1 + (long long)j * C;
  const float* s = sv + (long long)b * C;
  float a = 0.f;
  for (int c = 0; c < C; ++c) a += w[c] * s[c];
  hid[t] = a > 0.f ? a : 0.f;
}

__global__ void se2_scale_kernel(float* __restrict__ h, const float* __restrict__ hid,
                                 const float* __restrict__ w2,
                                 int B, int S, int C, int Ch) {
  int t = blockIdx.x * blockDim.x + threadIdx.x;
  if (t >= B * C) return;
  int b = t / C, c = t % C;
  const float* w = w2 + (long long)c * Ch;
  const float* hv = hid + (long long)b * Ch;
  float a = 0.f;
  for (int j = 0; j < Ch; ++j) a += w[j] * hv[j];
  a = 1.f / (1.f + expf(-a));
  for (int s = 0; s < S; ++s) h[((long long)b * S + s) * C + c] *= a;
}

__global__ void pool_kernel(const float* __restrict__ h, float* __restrict__ feat,
                            int B, int S, int C) {
  int t = blockIdx.x * blockDim.x + threadIdx.x;
  if (t >= B * C) return;
  int b = t / C, c = t % C;
  float mx = -3.0e38f, sm = 0.f;
  for (int s = 0; s < S; ++s) {
    float v = h[((long long)b * S + s) * C + c];
    mx = fmaxf(mx, v); sm += v;
  }
  feat[(long long)b * 2 * C + c] = mx;
  feat[(long long)b * 2 * C + C + c] = sm / (float)S;
}

// ---------------------------------------------------------------------------

extern "C" void kernel_launch(void* const* d_in, const int* in_sizes, int n_in,
                              void* d_out, int out_size, void* d_ws, size_t ws_size,
                              hipStream_t stream) {
  (void)n_in; (void)out_size; (void)ws_size;
  const int B = 16;
  static const int FD[4]   = {64, 128, 256, 512};
  static const int NP[4]   = {512, 256, 128, 64};
  static const int NIN[4]  = {2048, 512, 256, 128};
  static const int CGV[4]  = {6, 64, 128, 256};      // per-group edge input chans
  static const int KPE[4]  = {32, 64, 128, 256};     // edge K padded to 32
  static const int KPSA[4] = {96, 160, 288, 544};    // (C+3) padded to 32

  // ---- input index mapping (auto-detect flatten order via sizes) ----
  int xyz_i, rgb_i, p0;
  bool sorted_keys;
  if (in_sizes[0] == B * 2048 * 3) { xyz_i = 0; rgb_i = 1; p0 = 2;
                                     sorted_keys = (in_sizes[2] != 384); }
  else { p0 = 0; sorted_keys = (in_sizes[0] != 384); rgb_i = 61; xyz_i = 62; }
  int bnb[4], bng[4], convw[4], embw, embg, embb, projw, projb;
  int w1i[4], w2i[4], w3i[4], se1i[4], se2i[4];
  int g1i[4], b1i[4], g2i[4], b2i[4], g3i[4], b3i[4];
  if (sorted_keys) {
    for (int i = 0; i < 4; ++i) { bnb[i] = p0 + i; bng[i] = p0 + 4 + i; convw[i] = p0 + 8 + i; }
    embb = p0 + 12; embg = p0 + 13; embw = p0 + 14; projb = p0 + 15; projw = p0 + 16;
    for (int i = 0; i < 4; ++i) {
      int sb = p0 + 17 + 11 * i;
      b1i[i] = sb; b2i[i] = sb + 1; b3i[i] = sb + 2;
      g1i[i] = sb + 3; g2i[i] = sb + 4; g3i[i] = sb + 5;
      se1i[i] = sb + 6; se2i[i] = sb + 7;
      w1i[i] = sb + 8; w2i[i] = sb + 9; w3i[i] = sb + 10;
    }
  } else {
    for (int i = 0; i < 4; ++i) { convw[i] = p0 + i; bng[i] = p0 + 4 + i; bnb[i] = p0 + 8 + i; }
    for (int i = 0; i < 4; ++i) {
      int sb = p0 + 12 + 11 * i;
      w1i[i] = sb; w2i[i] = sb + 1; w3i[i] = sb + 2;
      se1i[i] = sb + 3; se2i[i] = sb + 4;
      g1i[i] = sb + 5; b1i[i] = sb + 6; g2i[i] = sb + 7; b2i[i] = sb + 8;
      g3i[i] = sb + 9; b3i[i] = sb + 10;
    }
    embw = p0 + 56; embg = p0 + 57; embb = p0 + 58; projw = p0 + 59; projb = p0 + 60;
  }
  const float* xyz_in = (const float*)d_in[xyz_i];
  const float* rgb_in = (const float*)d_in[rgb_i];
  auto F = [&](int i) { return (const float*)d_in[i]; };

  // ---- workspace arena ----
  char* ws = (char*)d_ws;
  size_t off = 0;
  auto alloc = [&](size_t bytes) -> void* {
    off = (off + 255) & ~(size_t)255;
    void* p = ws + off;
    off += bytes;
    return p;
  };
  unsigned short *convbf[4], *w1bf[4], *w2bf[4], *w3bf[4];
  for (int i = 0; i < 4; ++i) convbf[i] = (unsigned short*)alloc((size_t)FD[i] * KPE[i] * 2);
  for (int i = 0; i < 4; ++i) w1bf[i] = (unsigned short*)alloc((size_t)FD[i] * KPSA[i] * 2);
  for (int i = 0; i < 4; ++i) w2bf[i] = (unsigned short*)alloc((size_t)FD[i] * FD[i] * 2);
  for (int i = 0; i < 4; ++i) w3bf[i] = (unsigned short*)alloc((size_t)FD[i] * FD[i] * 2);
  unsigned short* embbf  = (unsigned short*)alloc((size_t)512 * 1024 * 2);
  unsigned short* projbf = (unsigned short*)alloc((size_t)751 * 512 * 2);
  float* hA   = (float*)alloc((size_t)B * 2048 * 64 * 4);   // edge outputs (B,N,C)
  float* hB   = (float*)alloc((size_t)B * 2048 * 64 * 4);   // SA outputs   (B,S,C)
  float* xyzA = (float*)alloc((size_t)B * 512 * 3 * 4);
  float* xyzB = (float*)alloc((size_t)B * 512 * 3 * 4);
  int* knnidx = (int*)alloc((size_t)B * 2048 * EDGE_K * 4);
  int* fidx   = (int*)alloc((size_t)B * 512 * 4);
  int* gidx   = (int*)alloc((size_t)B * 512 * NSAMP * 4);
  float* stats = (float*)alloc((size_t)2 * 512 * 4);
  float* svec  = (float*)alloc((size_t)B * 512 * 4);
  float* hid   = (float*)alloc((size_t)B * 128 * 4);
  float* feat  = (float*)alloc((size_t)B * 1024 * 4);
  unsigned short* featbf = (unsigned short*)alloc((size_t)B * 1024 * 2);
  float* ebuf  = (float*)alloc((size_t)B * 512 * 4);
  unsigned short* ebf = (unsigned short*)alloc((size_t)B * 512 * 2);
  char* big = (char*)alloc((size_t)240 << 20);
  const size_t EOUT_OFF = (size_t)44 << 20;   // edge stage: e | eout
  const size_t BUFA_OFF = (size_t)104 << 20;  // SA stage: xsa | bufA | bufB
  const size_t BUFB_OFF = (size_t)172 << 20;

  // ---- generic GEMM launcher (wave strip = 16M x 32N, block = 32M x 64N) ----
  auto gemm = [&](const unsigned short* A, const unsigned short* Bm,
                  float* Df, unsigned short* Dbf, const float* bias,
                  int M, int P, int K, int ldb, int ldd,
                  long long sB, long long sD, int batches) {
    dim3 grid(CDIV(M, 32), CDIV(P, 64), batches);
    if (Dbf)
      gemm_kernel<1><<<grid, 128, 0, stream>>>(A, Bm, nullptr, Dbf, bias,
                                               M, P, K, ldb, ldd, sB, sD);
    else
      gemm_kernel<0><<<grid, 128, 0, stream>>>(A, Bm, Df, nullptr, bias,
                                               M, P, K, ldb, ldd, sB, sD);
  };

  // ---- convert weights to padded bf16 ----
  for (int i = 0; i < 4; ++i) {
    cast_pad_kernel<<<CDIV(FD[i] * KPE[i], 256), 256, 0, stream>>>(
        F(convw[i]), convbf[i], FD[i], CGV[i], KPE[i]);
    cast_pad_kernel<<<CDIV(FD[i] * KPSA[i], 256), 256, 0, stream>>>(
        F(w1i[i]), w1bf[i], FD[i], FD[i] + 3, KPSA[i]);
    cast_pad_kernel<<<CDIV(FD[i] * FD[i], 256), 256, 0, stream>>>(
        F(w2i[i]), w2bf[i], FD[i], FD[i], FD[i]);
    cast_pad_kernel<<<CDIV(FD[i] * FD[i], 256), 256, 0, stream>>>(
        F(w3i[i]), w3bf[i], FD[i], FD[i], FD[i]);
  }
  cast_pad_kernel<<<CDIV(512 * 1024, 256), 256, 0, stream>>>(F(embw), embbf, 512, 1024, 1024);
  cast_pad_kernel<<<CDIV(751 * 512, 256), 256, 0, stream>>>(F(projw), projbf, 751, 512, 512);

  // ---- layer loop ----
  const float* xyz_cur = xyz_in;
  const float* h_cur = rgb_in;
  int Cin = 3;
  for (int i = 0; i < 4; ++i) {
    int N = NIN[i], C = FD[i], S = NP[i];
    int G = (i == 0) ? 1 : 2;
    int KpE = KPE[i], Lrow = G * KpE, P = N * EDGE_K;

    // --- EdgeConv ---
    knn_kernel<<<CDIV(B * N, 256), 256, 0, stream>>>(xyz_cur, knnidx, B, N);
    unsigned short* e = (unsigned short*)big;
    unsigned short* eout = (unsigned short*)(big + EOUT_OFF);
    edge_build_kernel<<<CDIV(B * N * EDGE_K, 256), 256, 0, stream>>>(
        h_cur, knnidx, e, B, N, Cin, G, KpE);
    int Coutg = C / G;
    for (int g = 0; g < G; ++g)
      gemm(convbf[i] + (size_t)g * Coutg * KpE, e + g * KpE,
           nullptr, eout + g * Coutg, nullptr,
           Coutg, P, KpE, Lrow, C, (long long)P * Lrow, (long long)P * C, B);
    long long tot = (long long)B * N * C;
    maxk_kernel<<<(unsigned)CDIV(tot, 256), 256, 0, stream>>>(eout, hA, B, N, C, EDGE_K);
    bn_stats_kernel<<<C, 256, 0, stream>>>(hA, nullptr, stats, B * N, C);
    bn_apply_kernel<<<(unsigned)CDIV(tot, 256), 256, 0, stream>>>(
        hA, nullptr, nullptr, nullptr, hA, nullptr,
        F(bng[i]), F(bnb[i]), stats, tot, C, /*leaky*/2);

    // --- SA module ---
    float* cen = (i & 1) ? xyzB : xyzA;
    fps_kernel<<<B, 256, 0, stream>>>(xyz_cur, fidx, cen, N, S);
    ball_kernel<<<CDIV(B * S, 256), 256, 0, stream>>>(xyz_cur, cen, gidx, B, N, S);
    int KpSA = KPSA[i], Psa = S * NSAMP;
    unsigned short* xsa  = (unsigned short*)big;
    unsigned short* bufA = (unsigned short*)(big + BUFA_OFF);
    unsigned short* bufB = (unsigned short*)(big + BUFB_OFF);
    group_build_kernel<<<CDIV(B * Psa, 256), 256, 0, stream>>>(
        xyz_cur, cen, hA, gidx, xsa, B, N, S, C, KpSA);
    long long tsa = (long long)B * Psa * C;
    // h1 = relu(bn(W1 x))
    gemm(w1bf[i], xsa, nullptr, bufA, nullptr, C, Psa, KpSA, KpSA, C,
         (long long)Psa * KpSA, (long long)Psa * C, B);
    bn_stats_kernel<<<C, 256, 0, stream>>>(nullptr, bufA, stats, B * Psa, C);
    bn_apply_kernel<<<(unsigned)CDIV(tsa, 256), 256, 0, stream>>>(
        nullptr, bufA, nullptr, nullptr, nullptr, bufA,
        F(g1i[i]), F(b1i[i]), stats, tsa, C, 1);
    // h2 = relu(bn(W2 h1)) + h1
    gemm(w2bf[i], bufA, nullptr, bufB, nullptr, C, Psa, C, C, C,
         (long long)Psa * C, (long long)Psa * C, B);
    bn_stats_kernel<<<C, 256, 0, stream>>>(nullptr, bufB, stats, B * Psa, C);
    bn_apply_kernel<<<(unsigned)CDIV(tsa, 256), 256, 0, stream>>>(
        nullptr, bufB, nullptr, bufA, nullptr, bufB,
        F(g2i[i]), F(b2i[i]), stats, tsa, C, 1);
    // h3 = relu(bn(W3 h2)) + h2
    gemm(w3bf[i], bufB, nullptr, bufA, nullptr, C, Psa, C, C, C,
         (long long)Psa * C, (long long)Psa * C, B);
    bn_stats_kernel<<<C, 256, 0, stream>>>(nullptr, bufA, stats, B * Psa, C);
    bn_apply_kernel<<<(unsigned)CDIV(tsa, 256), 256, 0, stream>>>(
        nullptr, bufA, nullptr, bufB, nullptr, bufA,
        F(g3i[i]), F(b3i[i]), stats, tsa, C, 1);
    // max over nsample -> (B,S,C); SE scale
    long long tso = (long long)B * S * C;
    maxk_kernel<<<(unsigned)CDIV(tso, 256), 256, 0, stream>>>(bufA, hB, B, S, C, NSAMP);
    se_mean_kernel<<<CDIV(B * C, 256), 256, 0, stream>>>(hB, svec, B, S, C);
    se1_kernel<<<CDIV(B * (C / 4), 256), 256, 0, stream>>>(svec, F(se1i[i]), hid, B, C, C / 4);
    se2_scale_kernel<<<CDIV(B * C, 256), 256, 0, stream>>>(hB, hid, F(se2i[i]), B, S, C, C / 4);

    xyz_cur = cen; h_cur = hB; Cin = C;
  }

  // ---- head: pool, embedding + BN, projection ----
  pool_kernel<<<CDIV(B * 512, 256), 256, 0, stream>>>(hB, feat, B, 64, 512);
  cast_pad_kernel<<<CDIV(B * 1024, 256), 256, 0, stream>>>(feat, featbf, B, 1024, 1024);
  gemm(embbf, featbf, ebuf, nullptr, nullptr, 512, B, 1024, 1024, 512, 0, 0, 1);
  bn_stats_kernel<<<512, 256, 0, stream>>>(ebuf, nullptr, stats, B, 512);
  bn_apply_kernel<<<CDIV(B * 512, 256), 256, 0, stream>>>(
      ebuf, nullptr, nullptr, nullptr, ebuf, ebf,
      F(embg), F(embb), stats, (long long)B * 512, 512, 0);
  gemm(projbf, ebf, (float*)d_out, nullptr, F(projb), 751, B, 512, 512, 751, 0, 0, 1);
}
